// RobustTrustWrapper_49890340110405
// MI455X (gfx1250) — matrix operand order repaired and checked by reading, not compiled, not run on hardware
//
#include <hip/hip_runtime.h>
#include <math.h>

#define NDEV 2048
#define KNBR 32

typedef _Float16 f16;
typedef __attribute__((ext_vector_type(8)))  _Float16 v8h;
typedef __attribute__((ext_vector_type(16))) _Float16 v16h;
typedef __attribute__((ext_vector_type(8)))  float    v8f;

// ---------------- workspace layout (bytes) ----------------
// [0,              8 MB)   : Mf16   (2048x2048 f16, 0/1)
// [8 MB,          24 MB)   : count  (2048x2048 f32)
// [24 MB,   24 MB+512KB)   : MTbits (2048 rows x 64 u32; bit j of row c = c in nbr(j))
// [.. ,     +512KB)        : Mbits  (2048 rows x 64 u32; bit j of row i = j in nbr(i))
// then                     : flags  (2048 u32)
// then                     : ctr    (1 u32)
#define OFF_MF16   ((size_t)0)
#define OFF_COUNT  ((size_t)8  * 1024 * 1024)
#define OFF_MTBITS ((size_t)24 * 1024 * 1024)
#define OFF_MBITS  (OFF_MTBITS + (size_t)NDEV * 64 * 4)
#define OFF_FLAGS  (OFF_MBITS  + (size_t)NDEV * 64 * 4)
#define OFF_CTR    (OFF_FLAGS  + (size_t)NDEV * 4)

// ---------------- kernel 1: zero the bit/flag regions ----------------
__global__ void trust_init_ws(unsigned* __restrict__ mtbits,
                              unsigned* __restrict__ mbits,
                              unsigned* __restrict__ flags,
                              unsigned* __restrict__ ctr) {
  size_t tid    = (size_t)blockIdx.x * blockDim.x + threadIdx.x;
  size_t stride = (size_t)gridDim.x * blockDim.x;
  size_t nwords = (size_t)NDEV * 64;
  for (size_t w = tid; w < nwords; w += stride) {
    mtbits[w] = 0u;
    mbits[w]  = 0u;
  }
  if (tid < NDEV) flags[tid] = 0u;
  if (tid == 0)   *ctr = 0u;
}

// ---------------- kernel 2: build M (f16 0/1) and membership bitmasks ----------------
// grid = NDEV blocks, 256 threads; block b owns row b.
__global__ void trust_build_m(const int* __restrict__ nbr,
                              f16* __restrict__ mf16,
                              unsigned* __restrict__ mbits,
                              unsigned* __restrict__ mtbits) {
  int i = blockIdx.x;
  f16* row = mf16 + (size_t)i * NDEV;
  // zero the row (256 threads x v8h = 2048 halves)
  *(v8h*)(row + threadIdx.x * 8) = (v8h)0;
  __syncthreads();
  if (threadIdx.x < KNBR) {
    int c = nbr[(size_t)i * KNBR + threadIdx.x];
    row[c] = (f16)1.0f;
    atomicOr(&mbits[(size_t)i * 64 + (c >> 5)], 1u << (c & 31));       // j=c in nbr(i)
    atomicOr(&mtbits[(size_t)c * 64 + (i >> 5)], 1u << (i & 31));      // c in nbr(i) -> bit i of MT row c
  }
}

// ---------------- kernel 3: count = M * M^T via v_wmma_f32_16x16x32_f16 ----------------
// 16384 16x16 tiles; 8 waves per block -> 2048 blocks. One tile per wave, K-loop of 64.
__global__ void trust_count_wmma(const f16* __restrict__ mf16,
                                 float* __restrict__ cnt) {
  int lane = threadIdx.x & 31;
  int wave = threadIdx.x >> 5;
  int tile = blockIdx.x * 8 + wave;          // 0 .. 16383
  int tileI = tile >> 7;                     // 0 .. 127
  int tileJ = tile & 127;

  int arow = tileI * 16 + (lane & 15);
  int brow = tileJ * 16 + (lane & 15);
  const f16* abase = mf16 + (size_t)arow * NDEV;
  const f16* bbase = mf16 + (size_t)brow * NDEV;

  v8f acc = (v8f)0;
  for (int k0 = 0; k0 < NDEV; k0 += 32) {
    // A fragment: 16x32 f16. lanes 0-15 hold K = k0+0..15; lanes 16-31 hold K = k0+16..31.
    int aoff = (lane < 16) ? k0 : (k0 + 16);
    union { v16h v; v8h h[2]; } ua;
    ua.v = *(const v16h*)(abase + aoff);
    // B fragment: 32x16 f16, B[k][n] = M[tileJ*16+n][k]. lanes 0-15: K=k0..k0+15 (n=lane),
    // lanes 16-31: K=k0+16..k0+31 (n=lane-16). Contiguous 32B per lane.
    int boff = (lane < 16) ? k0 : (k0 + 16);
    v16h b = *(const v16h*)(bbase + boff);
    acc = __builtin_amdgcn_wmma_f32_16x16x32_f16(
        /*neg_a=*/false, ua.v, /*neg_b=*/false, b,
        /*c_mod=*/(short)0, acc, /*reuse_a=*/false, /*reuse_b=*/false);
  }
  // D layout: VGPR r = row M=r (lanes 0-15, N=lane) / M=8+r (lanes 16-31, N=lane-16)
  int m0 = (lane < 16) ? 0 : 8;
  int jcol = tileJ * 16 + (lane & 15);
#pragma unroll
  for (int r = 0; r < 8; ++r) {
    cnt[(size_t)(tileI * 16 + m0 + r) * NDEV + jcol] = acc[r];
  }
}

// ---------------- kernel 4: dataflow sequential scan ----------------
// Persistent blocks claim rows in increasing order; spin on per-row done flags.
__global__ void trust_scan(const float* __restrict__ dyn,
                           const float* __restrict__ stat,
                           const int* __restrict__ nbr,
                           const float* __restrict__ cnt,
                           const unsigned* __restrict__ mbits,
                           const unsigned* __restrict__ mtbits,
                           float* __restrict__ out,
                           unsigned* __restrict__ flags,
                           unsigned* __restrict__ ctr) {
  __shared__ int      s_row;
  __shared__ int      s_n;
  __shared__ int      s_c[KNBR];
  __shared__ float    s_a[KNBR];
  __shared__ int      t_c[KNBR];
  __shared__ float    t_a[KNBR];
  __shared__ int      t_keep[KNBR];
  __shared__ unsigned s_mask[KNBR][64];   // MTbits rows of the deps (8 KB)

  for (;;) {
    if (threadIdx.x == 0) s_row = (int)atomicAdd(ctr, 1u);
    __syncthreads();
    int i = s_row;
    if (i >= NDEV) break;

    // direct-trust coefficients for the neighbor list of row i
    if (threadIdx.x < KNBR) {
      int c = nbr[(size_t)i * KNBR + threadIdx.x];
      t_c[threadIdx.x]    = c;
      t_keep[threadIdx.x] = (c < i) ? 1 : 0;
      float comb = 0.7f * dyn[(size_t)i * NDEV + c] + 0.3f * stat[c];
      t_a[threadIdx.x] = tanhf(0.5f * (comb + 0.5f));
    }
    __syncthreads();
    if (threadIdx.x == 0) {           // deterministic compaction (list order)
      int m = 0;
      for (int t = 0; t < KNBR; ++t)
        if (t_keep[t]) { s_c[m] = t_c[t]; s_a[m] = t_a[t]; ++m; }
      s_n = m;
    }
    __syncthreads();
    int nd = s_n;

    // wait until all dependency rows (c < i) are published
    if ((int)threadIdx.x < nd) {
      unsigned* f = &flags[s_c[threadIdx.x]];
      while (__hip_atomic_load(f, __ATOMIC_ACQUIRE, __HIP_MEMORY_SCOPE_AGENT) == 0u)
        __builtin_amdgcn_s_sleep(2);
    }
    __syncthreads();

    // stage MT bitmask rows of the deps into LDS
    for (int w = threadIdx.x; w < nd * 64; w += blockDim.x) {
      int t = w >> 6;
      s_mask[t][w & 63] = mtbits[(size_t)s_c[t] * 64 + (w & 63)];
    }
    __syncthreads();

    const unsigned* mrow = &mbits[(size_t)i * 64];
    for (int j = threadIdx.x; j < NDEV; j += blockDim.x) {
      float nv  = tanhf(0.5f * (0.7f * dyn[(size_t)i * NDEV + j] + 0.3f * stat[j] + 0.5f));
      float cij = cnt[(size_t)i * NDEV + j];
      bool isnbr = (mrow[j >> 5] >> (j & 31)) & 1u;
      float val;
      if (j == i) {
        val = 1.0f;
      } else if (isnbr) {
        val = nv;
      } else if (cij > 0.0f) {
        float num = 0.0f;
        for (int t = 0; t < nd; ++t) {
          int c = s_c[t];
          unsigned mb = (s_mask[t][j >> 5] >> (j & 31)) & 1u;
          if (mb && (c < j))
            num = fmaf(s_a[t], out[(size_t)c * NDEV + j], num);
        }
        val = 0.8f * num / cij;       // cij >= 1 when > 0, so max(cij,1)==cij
      } else {
        val = 0.5f * stat[j];
      }
      val = fminf(fmaxf(val, 0.0f), 1.0f);
      out[(size_t)i * NDEV + j] = val;
    }

    __threadfence();                  // each thread publishes its stores (agent scope)
    __syncthreads();
    if (threadIdx.x == 0)
      __hip_atomic_store(&flags[i], 1u, __ATOMIC_RELEASE, __HIP_MEMORY_SCOPE_AGENT);
  }
}

// ---------------- launch ----------------
extern "C" void kernel_launch(void* const* d_in, const int* in_sizes, int n_in,
                              void* d_out, int out_size, void* d_ws, size_t ws_size,
                              hipStream_t stream) {
  const float* dyn  = (const float*)d_in[0];   // 2048*2048
  const float* stat = (const float*)d_in[1];   // 2048
  const int*   nbr  = (const int*)d_in[2];     // 2048*32
  float* out = (float*)d_out;

  char* ws = (char*)d_ws;
  f16*      mf16   = (f16*)(ws + OFF_MF16);
  float*    cnt    = (float*)(ws + OFF_COUNT);
  unsigned* mtbits = (unsigned*)(ws + OFF_MTBITS);
  unsigned* mbits  = (unsigned*)(ws + OFF_MBITS);
  unsigned* flags  = (unsigned*)(ws + OFF_FLAGS);
  unsigned* ctr    = (unsigned*)(ws + OFF_CTR);

  (void)in_sizes; (void)n_in; (void)out_size; (void)ws_size;

  trust_init_ws<<<512, 256, 0, stream>>>(mtbits, mbits, flags, ctr);
  trust_build_m<<<NDEV, 256, 0, stream>>>(nbr, mf16, mbits, mtbits);
  trust_count_wmma<<<2048, 256, 0, stream>>>(mf16, cnt);   // 8 waves/block, 1 tile/wave
  trust_scan<<<64, 256, 0, stream>>>(dyn, stat, nbr, cnt, mbits, mtbits, out, flags, ctr);
}
